// Attention_15109694947883
// MI455X (gfx1250) — matrix operand order repaired and checked by reading, not compiled, not run on hardware
//
#include <hip/hip_runtime.h>
#include <hip/hip_bf16.h>

// ---------------------------------------------------------------------------
// CDNA5 (gfx1250) wave32 WMMA attention:
//   K1: Q = X @ Wqkv[:, :1024]           (bf16 v_wmma_f32_16x16x32_bf16)
//   K2: flash causal attention, Q=K=V    (bf16 WMMA, f32 online softmax)
//   K3: out = A @ Wout + b               (bf16 WMMA, f32 epilogue)
// All fragment feeds are 128-bit (ds_load_b128 / global_load_b128); softmax
// row reductions use DPP quad_perm + v_permlane16 (no LDS bpermute); next
// tiles are prefetched (global_prefetch_b8) while the current tile computes.
// ---------------------------------------------------------------------------

typedef __attribute__((ext_vector_type(16))) __bf16 bf16x16;
typedef __attribute__((ext_vector_type(8)))  float  v8f;

#define DIM      1024
#define HEADS    16
#define DIM_HEAD 64
#define SEQ      2048
#define ROWS     4096        // 2 * 2048
#define SCALE    0.125f      // 64^-0.5

union B16x16 { bf16x16 v; uint4 q[2]; };
union B16x8  { uint4 q; __bf16 e[8]; };

__device__ __forceinline__ __bf16 f2bf(float f) {
    unsigned u = __builtin_bit_cast(unsigned, f);
    unsigned r = u + 0x7FFFu + ((u >> 16) & 1u);       // round-to-nearest-even
    return __builtin_bit_cast(__bf16, (unsigned short)(r >> 16));
}
__device__ __forceinline__ unsigned pack2(float a, float b) {
    unsigned ha = (unsigned)__builtin_bit_cast(unsigned short, f2bf(a));
    unsigned hb = (unsigned)__builtin_bit_cast(unsigned short, f2bf(b));
    return ha | (hb << 16);
}

__device__ __forceinline__ v8f zero8() {
    v8f z;
#pragma unroll
    for (int i = 0; i < 8; ++i) z[i] = 0.f;
    return z;
}

__device__ __forceinline__ v8f wmma_bf16(bf16x16 a, bf16x16 b, v8f c) {
    return __builtin_amdgcn_wmma_f32_16x16x32_bf16(false, a, false, b,
                                                   (short)0, c, false, false);
}

// ---- VALU cross-lane reductions within 16-lane rows (no LDS) ---------------
template <int CTRL>
__device__ __forceinline__ float dpp_mov(float x) {
    return __builtin_bit_cast(float,
        __builtin_amdgcn_update_dpp(0, __builtin_bit_cast(int, x),
                                    CTRL, 0xF, 0xF, true));
}
__device__ __forceinline__ float permlane16_f(float x, unsigned lo, unsigned hi) {
    int xi = __builtin_bit_cast(int, x);
    return __builtin_bit_cast(float,
        __builtin_amdgcn_permlane16(xi, xi, (int)lo, (int)hi, true, false));
}
__device__ __forceinline__ float red_max16(float x) {
    x = fmaxf(x, dpp_mov<0xB1>(x));                              // xor 1
    x = fmaxf(x, dpp_mov<0x4E>(x));                              // xor 2
    x = fmaxf(x, permlane16_f(x, 0x32107654u, 0xBA98FEDCu));     // xor 4
    x = fmaxf(x, permlane16_f(x, 0xFEDCBA98u, 0x76543210u));     // xor 8
    return x;
}
__device__ __forceinline__ float red_sum16(float x) {
    x += dpp_mov<0xB1>(x);
    x += dpp_mov<0x4E>(x);
    x += permlane16_f(x, 0x32107654u, 0xBA98FEDCu);
    x += permlane16_f(x, 0xFEDCBA98u, 0x76543210u);
    return x;
}

// ---- fragment loads (ISA 7.12.2 layouts), always 2 x 128-bit ---------------
__device__ __forceinline__ bf16x16 load_a16(const __bf16* tile, int stride, int lane) {
    const __bf16* p = tile + (lane & 15) * stride + ((lane >> 4) << 3);
    B16x16 r;
    r.q[0] = *(const uint4*)(p);        // K = kb .. kb+7
    r.q[1] = *(const uint4*)(p + 16);   // K = kb+16 .. kb+23
    return r.v;
}
__device__ __forceinline__ bf16x16 load_b16(const __bf16* tile, int nstride, int lane) {
    const __bf16* p = tile + (lane & 15) * nstride + ((lane >> 4) << 3);
    B16x16 r;
    r.q[0] = *(const uint4*)(p);
    r.q[1] = *(const uint4*)(p + 16);
    return r.v;
}

// ---------------------------------------------------------------------------
// GEMM: C[MxN] = A[MxK] * B[KxN] (+bias). Block 128x64, K-step 32, 8 waves.
// B is staged TRANSPOSED in LDS ([n][k]) so B-fragments are 2x ds_load_b128.
// ---------------------------------------------------------------------------
template <bool A_BF16, bool OUT_F32>
__global__ __launch_bounds__(256)
void gemm_bf16_wmma(const void* __restrict__ Ap, int lda,
                    const float* __restrict__ B, int ldb,
                    const float* __restrict__ bias,
                    void* __restrict__ Cp, int ldc, int K) {
    __shared__ alignas(16) __bf16 a_lds[128 * 32];   // [m][k]
    __shared__ alignas(16) __bf16 b_ldsT[64 * 32];   // [n][k]

    const int tid  = threadIdx.x;
    const int wave = tid >> 5;
    const int lane = tid & 31;
    const int m0   = blockIdx.y * 128;
    const int n0   = blockIdx.x * 64;

    v8f acc[4];
#pragma unroll
    for (int i = 0; i < 4; ++i) acc[i] = zero8();

    for (int k0 = 0; k0 < K; k0 += 32) {
        __syncthreads();
        // ---- stage A (128x32) ----
        if (A_BF16) {
#pragma unroll
            for (int i = 0; i < 2; ++i) {
                int idx4 = tid + i * 256;
                int r = idx4 >> 2, c4 = idx4 & 3;
                ((uint4*)a_lds)[idx4] =
                    *((const uint4*)((const __bf16*)Ap + (size_t)(m0 + r) * lda + k0) + c4);
            }
        } else {
#pragma unroll
            for (int i = 0; i < 4; ++i) {
                int idx4 = tid + i * 256;
                int r = idx4 >> 3, c4 = idx4 & 7;
                float4 f = ((const float4*)((const float*)Ap + (size_t)(m0 + r) * lda + k0))[c4];
                uint2 p2; p2.x = pack2(f.x, f.y); p2.y = pack2(f.z, f.w);
                *(uint2*)(a_lds + r * 32 + c4 * 4) = p2;
            }
        }
        // ---- stage B transposed (32x64 -> [n][k]) ----
#pragma unroll
        for (int i = 0; i < 2; ++i) {
            int idx4 = tid + i * 256;
            int r = idx4 >> 4, c4 = idx4 & 15;
            float4 f = ((const float4*)(B + (size_t)(k0 + r) * ldb + n0))[c4];
            int cb = c4 * 4;
            b_ldsT[(cb + 0) * 32 + r] = f2bf(f.x);
            b_ldsT[(cb + 1) * 32 + r] = f2bf(f.y);
            b_ldsT[(cb + 2) * 32 + r] = f2bf(f.z);
            b_ldsT[(cb + 3) * 32 + r] = f2bf(f.w);
        }
        // prefetch next K-step tiles (uniform guard; one line per thread)
        if (k0 + 32 < K) {
            int ra = tid >> 1;                                  // 128 rows, 2 lines
            if (A_BF16)
                __builtin_prefetch((const __bf16*)Ap + (size_t)(m0 + ra) * lda + (k0 + 32)
                                       + (tid & 1) * 16, 0, 0);
            else
                __builtin_prefetch((const float*)Ap + (size_t)(m0 + ra) * lda + (k0 + 32)
                                       + (tid & 1) * 16, 0, 0);
            int rb = tid >> 3;                                  // 32 rows, 8 lines
            __builtin_prefetch(B + (size_t)(k0 + 32 + rb) * ldb + n0 + (tid & 7) * 8, 0, 0);
        }
        __syncthreads();

        bf16x16 af = load_a16(a_lds + wave * 16 * 32, 32, lane);
#pragma unroll
        for (int nc = 0; nc < 4; ++nc) {
            bf16x16 bf = load_b16(b_ldsT + nc * 16 * 32, 32, lane);
            acc[nc] = wmma_bf16(af, bf, acc[nc]);
        }
    }

    const int nl = lane & 15, hi = lane >> 4;
    float bv[4];
#pragma unroll
    for (int nc = 0; nc < 4; ++nc)
        bv[nc] = (OUT_F32 && bias) ? bias[n0 + nc * 16 + nl] : 0.f;
#pragma unroll
    for (int nc = 0; nc < 4; ++nc) {
#pragma unroll
        for (int v = 0; v < 8; ++v) {
            int row = m0 + wave * 16 + v + 8 * hi;
            int col = n0 + nc * 16 + nl;
            float x = acc[nc][v];
            if (OUT_F32) ((float*)Cp)[(size_t)row * ldc + col] = x + bv[nc];
            else         ((__bf16*)Cp)[(size_t)row * ldc + col] = f2bf(x);
        }
    }
}

// ---------------------------------------------------------------------------
// Flash causal attention, Q=K=V = qb. 1 workgroup = 4 waves = 64 query rows
// of one (b,h). KV tile staged row-major (scores) + transposed (P.V).
// ---------------------------------------------------------------------------
__global__ __launch_bounds__(128)
void flash_attn_wmma(const void* __restrict__ qbv, void* __restrict__ aov) {
    const __bf16* qb = (const __bf16*)qbv;
    __bf16*       ao = (__bf16*)aov;

    const int qt   = blockIdx.x;
    const int h    = blockIdx.y;
    const int bt   = blockIdx.z;
    const int wave = threadIdx.x >> 5;
    const int lane = threadIdx.x & 31;
    const int nl   = lane & 15;
    const int hi   = lane >> 4;

    __shared__ alignas(16) __bf16 kv [32 * DIM_HEAD];   // [j][d]
    __shared__ alignas(16) __bf16 kvT[DIM_HEAD * 32];   // [d][j]
    __shared__ alignas(16) __bf16 pst[4][16 * 32];      // per-wave P relayout

    const int row_base = bt * SEQ;
    const int q0w  = qt * 64 + wave * 16;
    const int col0 = h * DIM_HEAD;

    // Q fragments (A layout) straight from global: 4x global_load_b128
    bf16x16 qf0 = load_a16(qb + (size_t)(row_base + q0w) * DIM + col0,      DIM, lane);
    bf16x16 qf1 = load_a16(qb + (size_t)(row_base + q0w) * DIM + col0 + 32, DIM, lane);

    v8f o0 = zero8(), o1 = zero8(), o2 = zero8(), o3 = zero8();
    float m_r[8], l_r[8];
#pragma unroll
    for (int v = 0; v < 8; ++v) { m_r[v] = -1e30f; l_r[v] = 0.f; }

    const int ktiles = 2 * qt + 2;
    for (int kt = 0; kt < ktiles; ++kt) {
        const int k0 = kt * 32;
        __syncthreads();
        // ---- stage KV tile (b128 loads) + transposed copy ----
        uint4* kv4 = (uint4*)kv;
#pragma unroll
        for (int i = 0; i < 2; ++i) {                  // 256 uint4
            int idx = threadIdx.x + i * 128;
            int j = idx >> 3, c4 = idx & 7;
            B16x8 u;
            u.q = ((const uint4*)(qb + (size_t)(row_base + k0 + j) * DIM + col0))[c4];
            kv4[idx] = u.q;
            int c0 = c4 * 8;
#pragma unroll
            for (int t = 0; t < 8; ++t) kvT[(c0 + t) * 32 + j] = u.e[t];
        }
        // prefetch next KV tile while this one computes (uniform guard)
        if (kt + 1 < ktiles) {
            int j = threadIdx.x >> 2;                  // 32 rows, 4 x 32B lines
            __builtin_prefetch(qb + (size_t)(row_base + k0 + 32 + j) * DIM + col0
                                   + (threadIdx.x & 3) * 16, 0, 0);
        }
        __syncthreads();

        // ---- S = Q K^T (contraction D=64), two 16-key tiles ----
        v8f s[2];
#pragma unroll
        for (int nh = 0; nh < 2; ++nh) {
            const __bf16* base = kv + nh * 16 * DIM_HEAD;
            v8f a = zero8();
            a = wmma_bf16(qf0, load_b16(base,      DIM_HEAD, lane), a);
            a = wmma_bf16(qf1, load_b16(base + 32, DIM_HEAD, lane), a);
            s[nh] = a;
        }

        // ---- scale + causal mask (only near diagonal) ----
        float sc0[8], sc1[8];
#pragma unroll
        for (int v = 0; v < 8; ++v) {
            sc0[v] = s[0][v] * SCALE;
            sc1[v] = s[1][v] * SCALE;
        }
        if (k0 + 31 > q0w) {                           // wave-uniform branch
#pragma unroll
            for (int v = 0; v < 8; ++v) {
                const int qi = q0w + v + 8 * hi;
                if (k0 + nl      > qi) sc0[v] = -1e30f;
                if (k0 + 16 + nl > qi) sc1[v] = -1e30f;
            }
        }

        // ---- online softmax (VALU-only reductions) ----
        float p0[8], p1[8], corr[8];
#pragma unroll
        for (int v = 0; v < 8; ++v) {
            float r  = red_max16(fmaxf(sc0[v], sc1[v]));
            float mn = fmaxf(m_r[v], r);
            float c  = __expf(m_r[v] - mn);
            float e0 = __expf(sc0[v] - mn);
            float e1 = __expf(sc1[v] - mn);
            float rs = red_sum16(e0 + e1);

            l_r[v] = l_r[v] * c + rs;
            m_r[v] = mn;
            corr[v] = c;
            p0[v] = e0; p1[v] = e1;
        }
#pragma unroll
        for (int v = 0; v < 8; ++v) {
            o0[v] *= corr[v]; o1[v] *= corr[v];
            o2[v] *= corr[v]; o3[v] *= corr[v];
        }

        // ---- P: C layout -> row-major 16x32 in LDS, reload in A layout ----
        __bf16* pw = &pst[wave][0];
#pragma unroll
        for (int v = 0; v < 8; ++v) {
            int m = v + 8 * hi;
            pw[m * 32 + nl]      = f2bf(p0[v]);
            pw[m * 32 + 16 + nl] = f2bf(p1[v]);
        }
        __syncthreads();

        // ---- O += P (16x32) * V (32x64); B frags from transposed kvT ----
        bf16x16 pa = load_a16(pw, 32, lane);
        o0 = wmma_bf16(pa, load_b16(kvT +  0 * 32, 32, lane), o0);
        o1 = wmma_bf16(pa, load_b16(kvT + 16 * 32, 32, lane), o1);
        o2 = wmma_bf16(pa, load_b16(kvT + 32 * 32, 32, lane), o2);
        o3 = wmma_bf16(pa, load_b16(kvT + 48 * 32, 32, lane), o3);
    }

    // ---- finalize: divide by l, store bf16 attention output ----
#pragma unroll
    for (int v = 0; v < 8; ++v) {
        float inv = 1.0f / l_r[v];
        int row = row_base + q0w + v + 8 * hi;
        __bf16* orow = ao + (size_t)row * DIM + col0;
        orow[ 0 + nl] = f2bf(o0[v] * inv);
        orow[16 + nl] = f2bf(o1[v] * inv);
        orow[32 + nl] = f2bf(o2[v] * inv);
        orow[48 + nl] = f2bf(o3[v] * inv);
    }
}

// ---------------------------------------------------------------------------
extern "C" void kernel_launch(void* const* d_in, const int* in_sizes, int n_in,
                              void* d_out, int out_size, void* d_ws, size_t ws_size,
                              hipStream_t stream) {
    const float* x     = (const float*)d_in[0];   // (2,2048,1024)
    const float* w_qkv = (const float*)d_in[1];   // (1024,3072); only cols 0..1023 live
    const float* w_out = (const float*)d_in[2];   // (1024,1024)
    const float* b_out = (const float*)d_in[3];   // (1024,)
    float* out = (float*)d_out;

    void* qb = d_ws;                                        // 4096x1024 bf16
    void* ao = (void*)((char*)d_ws + (size_t)ROWS * DIM * 2);

    dim3 gblk(256);
    dim3 ggrid(DIM / 64, ROWS / 128);

    gemm_bf16_wmma<false, false><<<ggrid, gblk, 0, stream>>>(
        x, DIM, w_qkv, 3 * DIM, nullptr, qb, DIM, DIM);

    flash_attn_wmma<<<dim3(SEQ / 64, HEADS, 2), 128, 0, stream>>>(qb, ao);

    gemm_bf16_wmma<true, true><<<ggrid, gblk, 0, stream>>>(
        ao, DIM, w_out, DIM, b_out, out, DIM, DIM);
}
